// FFM_73907797229839
// MI455X (gfx1250) — matrix operand order repaired and checked by reading, not compiled, not run on hardware
//
#include <hip/hip_runtime.h>

// ---------------- problem constants ----------------
#define NUM_FIELDS 20
#define FIELD_SIZE 500
#define SP 512                 // S padded to a multiple of 32 (WMMA K)
#define KDIM 16
#define TOTAL 10000            // NUM_FIELDS * FIELD_SIZE
#define BATCH 4096
#define PAIRS 190              // C(20,2)
#define PAIR_CHUNKS 10
#define PAIRS_PER_CHUNK 19     // 10 * 19 == 190
#define WG_ROWS 64             // batch rows per workgroup
#define WG_THREADS 128         // 4 waves (wave32); one 16-row subtile per wave
#define ROW_TILES (BATCH / WG_ROWS)  // 64

// workspace (bf16, pre-padded) layout
#define XB_ROW_STRIDE (NUM_FIELDS * SP)          // 10240 bf16 per batch row
#define XB_ELEMS ((size_t)BATCH * XB_ROW_STRIDE) // 41.9M bf16 = 83.9 MB
#define VB_SLAB (SP * KDIM)                      // 8192 bf16 = 16 KB per (g,fld)
#define VB_ELEMS ((size_t)NUM_FIELDS * NUM_FIELDS * VB_SLAB)

typedef __attribute__((ext_vector_type(16))) __bf16 v16bf;
typedef __attribute__((ext_vector_type(8)))  float  v8f;

union FragAB { uint4 q[2]; v16bf v; };

// A fragment (16x32 bf16, M x K): lane (row = lane&15, hi = lane>>4) needs two
// contiguous 8-element chunks: K in [kb+8*hi, +8) and [kb+16+8*hi, +8).
__device__ __forceinline__ v16bf load_a_frag(const __bf16* base, int row, int kb, int hi) {
  FragAB f;
  const __bf16* p = base + row * SP + kb + 8 * hi;
  f.q[0] = *reinterpret_cast<const uint4*>(p);
  f.q[1] = *reinterpret_cast<const uint4*>(p + 16);
  return f.v;
}

// B fragment (32x16 bf16, K x N): lane (n = lane&15, hi = lane>>4) holds column n,
// K in [kb+8*hi, +8) then [kb+16+8*hi, +8).
__device__ __forceinline__ v16bf load_b_frag(const __bf16* vbase, int kb, int lane) {
  FragAB f;
  const __bf16* p = vbase + (kb + 8 * (lane >> 4)) * KDIM + (lane & 15);
  #pragma unroll
  for (int s = 0; s < 8; ++s) {
    f.v[s] = p[s * KDIM];
    f.v[8 + s] = p[(16 + s) * KDIM];
  }
  return f.v;
}

// ---------------- async global->LDS staging (CDNA5, tracked by ASYNCcnt) ----
__device__ __forceinline__ void async_b128(unsigned ldsByteOff, unsigned vByteOff,
                                           unsigned long long sbase) {
  asm volatile("global_load_async_to_lds_b128 %0, %1, %2"
               :: "v"(ldsByteOff), "v"(vByteOff), "s"(sbase) : "memory");
}
__device__ __forceinline__ void wait_async_all() { asm volatile("s_wait_asynccnt 0x0"  ::: "memory"); }
__device__ __forceinline__ void wait_async_k48() { asm volatile("s_wait_asynccnt 0x30" ::: "memory"); }

// Stage one pre-converted x field tile: WG_ROWS x SP bf16 (32 instr / wave).
__device__ __forceinline__ void issue_x_async(unsigned ldsOff, const __bf16* xb,
                                              int rowbase, int fld, int tid) {
  const unsigned long long sbase = (unsigned long long)(uintptr_t)
      (xb + (size_t)rowbase * XB_ROW_STRIDE + (size_t)fld * SP);
  for (int idx = tid; idx < WG_ROWS * (SP / 8); idx += WG_THREADS) {  // 4096 b128
    int r = idx >> 6, c = idx & 63;                                   // 64 b128 / row
    async_b128(ldsOff + (unsigned)(r * (SP * 2) + c * 16),
               (unsigned)(r * (XB_ROW_STRIDE * 2) + c * 16), sbase);
  }
}

// Stage one pre-converted V slab: SP x KDIM bf16, contiguous (8 instr / wave).
__device__ __forceinline__ void issue_v_async(unsigned ldsOff, const __bf16* vb,
                                              int g, int fld, int tid) {
  const unsigned long long sbase = (unsigned long long)(uintptr_t)
      (vb + (size_t)(g * NUM_FIELDS + fld) * VB_SLAB);
  for (int idx = tid; idx < VB_SLAB / 8; idx += WG_THREADS) {         // 1024 b128
    unsigned off = (unsigned)(idx * 16);
    async_b128(ldsOff + off, off, sbase);
  }
}

// ---------------- one-time f32 -> bf16 pre-conversion into workspace --------
// xb[row][fld][0..511] (cols 500..511 zero)
__global__ __launch_bounds__(256) void convert_x_kernel(const float* __restrict__ x,
                                                        __bf16* __restrict__ xb) {
  const int row = blockIdx.x;
  const float* src = x + (size_t)row * TOTAL;
  __bf16* dst = xb + (size_t)row * XB_ROW_STRIDE;
  for (int idx = threadIdx.x; idx < XB_ROW_STRIDE / 4; idx += 256) {
    int fld = idx >> 7;                  // 128 quads per 512-col field
    int s4  = (idx & 127) * 4;
    alignas(8) __bf16 h[4] = {};
    if (s4 < FIELD_SIZE) {               // FIELD_SIZE % 4 == 0
      float4 f = *reinterpret_cast<const float4*>(src + fld * FIELD_SIZE + s4);
      h[0] = (__bf16)f.x; h[1] = (__bf16)f.y; h[2] = (__bf16)f.z; h[3] = (__bf16)f.w;
    }
    *reinterpret_cast<uint2*>(dst + 4 * idx) = *reinterpret_cast<const uint2*>(h);
  }
}

// vb[(g*20+fld)][0..511][16] (rows 500..511 zero)
__global__ __launch_bounds__(256) void convert_v_kernel(const float* __restrict__ v,
                                                        __bf16* __restrict__ vb) {
  const int slab = blockIdx.x;           // g*NUM_FIELDS + fld
  const int g = slab / NUM_FIELDS, fld = slab % NUM_FIELDS;
  const float* src = v + ((size_t)g * TOTAL + (size_t)fld * FIELD_SIZE) * KDIM;
  __bf16* dst = vb + (size_t)slab * VB_SLAB;
  for (int idx = threadIdx.x; idx < VB_SLAB / 4; idx += 256) {        // 2048 quads
    alignas(8) __bf16 h[4] = {};
    if (idx < FIELD_SIZE * KDIM / 4) {
      float4 f = reinterpret_cast<const float4*>(src)[idx];
      h[0] = (__bf16)f.x; h[1] = (__bf16)f.y; h[2] = (__bf16)f.z; h[3] = (__bf16)f.w;
    }
    *reinterpret_cast<uint2*>(dst + 4 * idx) = *reinterpret_cast<const uint2*>(h);
  }
}

// ---------------- linear term: out[b] = x[b,:] . w (f32, memory-bound) ------
__global__ __launch_bounds__(256) void ffm_linear_kernel(const float* __restrict__ x,
                                                         const float* __restrict__ w,
                                                         float* __restrict__ out) {
  __shared__ float red[8];
  const int b = blockIdx.x;
  const float* xr = x + (size_t)b * TOTAL;
  float s = 0.f;
  for (int c = threadIdx.x; c < TOTAL / 4; c += 256) {
    float4 xv = reinterpret_cast<const float4*>(xr)[c];
    float4 wv = reinterpret_cast<const float4*>(w)[c];
    s = fmaf(xv.x, wv.x, s); s = fmaf(xv.y, wv.y, s);
    s = fmaf(xv.z, wv.z, s); s = fmaf(xv.w, wv.w, s);
  }
  for (int off = 16; off >= 1; off >>= 1) s += __shfl_xor(s, off, 32);
  if ((threadIdx.x & 31) == 0) red[threadIdx.x >> 5] = s;
  __syncthreads();
  if (threadIdx.x < 8) {
    float t = red[threadIdx.x];
    for (int off = 4; off >= 1; off >>= 1) t += __shfl_xor(t, off, 8);
    if (threadIdx.x == 0) out[b] = t;   // initializes out (pair kernel adds)
  }
}

// ---------------- one pair iteration (fixed buffers -> constant LDS offsets)
// Compute pair (i,j) from (XjC, VjiC, VijC); prefetch pair (ni,nj) into the
// *other* buffer (xjOffN, vjiOffN, vijOffN). Advances (i, j).
__device__ __forceinline__ void pair_iter(
    bool has_next,
    const __bf16* Xi, const __bf16* XjC, const __bf16* VjiC, const __bf16* VijC,
    unsigned xiOff, unsigned xjOffN, unsigned vjiOffN, unsigned vijOffN,
    const __bf16* xb, const __bf16* vb, int rowbase,
    int tid, int wave, int row, int hi, int lane,
    int& i, int& j, int& cur_i, v8f& acc)
{
  // Xi reload (single-buffered; previous iteration's tail barrier has passed,
  // and the ASYNCcnt wait below covers this batch before compute).
  if (i != cur_i) { issue_x_async(xiOff, xb, rowbase, i, tid); cur_i = i; }

  int ni = i, nj = j + 1;
  if (nj == NUM_FIELDS) { ni = i + 1; nj = ni + 1; }

  if (has_next) {                        // prefetch next pair (48 asyncs/wave)
    issue_x_async(xjOffN, xb, rowbase, nj, tid);
    issue_v_async(vjiOffN, vb, nj, ni, tid);
    issue_v_async(vijOffN, vb, ni, nj, tid);
    wait_async_k48();                    // in-order: batch(p)+Xi done, next flies
  } else {
    wait_async_all();
  }
  __syncthreads();                       // staged data visible to all waves

  const __bf16* XiW = Xi  + wave * 16 * SP;
  const __bf16* XjW = XjC + wave * 16 * SP;

  // E1 = X_i(16x512) @ V_ji(512x16), E2 = X_j @ V_ij (interleaved chains)
  v8f e1 = {}, e2 = {};
  #pragma unroll
  for (int t = 0; t < SP / 32; ++t) {
    const int kb = 32 * t;
    v16bf ai = load_a_frag(XiW, row, kb, hi);
    v16bf bj = load_b_frag(VjiC, kb, lane);
    v16bf aj = load_a_frag(XjW, row, kb, hi);
    v16bf bi = load_b_frag(VijC, kb, lane);
    e1 = __builtin_amdgcn_wmma_f32_16x16x32_bf16(false, ai, false, bj,
                                                 (short)0, e1, false, false);
    e2 = __builtin_amdgcn_wmma_f32_16x16x32_bf16(false, aj, false, bi,
                                                 (short)0, e2, false, false);
  }
  acc += e1 * e2;                        // elementwise; dot over k done later

  __syncthreads();                       // done reading before buffer refills
  i = ni; j = nj;
}

// ---------------- pairwise term ---------------------------------------------
// grid.x = row tile (64 rows), grid.y = chunk of 19 (i<j) pairs.
// Double-buffered async staging, pair loop unrolled by 2 so each body uses
// compile-time-fixed LDS buffers (no per-iteration cselect address math).
__global__ __launch_bounds__(WG_THREADS) void ffm_pair_kernel(const __bf16* __restrict__ xb,
                                                              const __bf16* __restrict__ vb,
                                                              float* __restrict__ out) {
  extern __shared__ __align__(16) char smem[];
  __bf16* Xi   = reinterpret_cast<__bf16*>(smem);      // [64][512]      64 KB
  __bf16* Xj0  = Xi  + WG_ROWS * SP;                   // [64][512]      64 KB
  __bf16* Xj1  = Xj0 + WG_ROWS * SP;                   // [64][512]      64 KB
  __bf16* Vji0 = Xj1 + WG_ROWS * SP;                   // [512][16]      16 KB
  __bf16* Vji1 = Vji0 + VB_SLAB;                       //                16 KB
  __bf16* Vij0 = Vji1 + VB_SLAB;                       //                16 KB
  __bf16* Vij1 = Vij0 + VB_SLAB;                       //                16 KB -> 256 KB

  const unsigned xiOff   = (unsigned)(uintptr_t)Xi;    // low 32 bits of the
  const unsigned xjOff0  = (unsigned)(uintptr_t)Xj0;   // generic pointer ==
  const unsigned xjOff1  = (unsigned)(uintptr_t)Xj1;   // wave-relative LDS addr
  const unsigned vjiOff0 = (unsigned)(uintptr_t)Vji0;
  const unsigned vjiOff1 = (unsigned)(uintptr_t)Vji1;
  const unsigned vijOff0 = (unsigned)(uintptr_t)Vij0;
  const unsigned vijOff1 = (unsigned)(uintptr_t)Vij1;

  const int tid  = threadIdx.x;
  const int lane = tid & 31;
  const int wave = tid >> 5;
  const int row  = lane & 15;
  const int hi   = lane >> 4;
  const int rowbase = blockIdx.x * WG_ROWS;

  // decode this chunk's first pair (i, j), i < j
  int i = 0, rem = blockIdx.y * PAIRS_PER_CHUNK;
  while (rem >= NUM_FIELDS - 1 - i) { rem -= NUM_FIELDS - 1 - i; ++i; }
  int j = i + 1 + rem;

  // prologue: stage pair 0 into buffer 0
  issue_x_async(xjOff0, xb, rowbase, j, tid);
  issue_v_async(vjiOff0, vb, j, i, tid);
  issue_v_async(vijOff0, vb, i, j, tid);

  v8f acc = {};        // per-lane rows: M = q + 8*hi, summed over pairs and k
  int cur_i = -1;

  #pragma unroll 1
  for (int p = 0; p < PAIRS_PER_CHUNK; p += 2) {
    // even iteration: compute buf0, prefetch into buf1
    pair_iter(p + 1 < PAIRS_PER_CHUNK,
              Xi, Xj0, Vji0, Vij0, xiOff, xjOff1, vjiOff1, vijOff1,
              xb, vb, rowbase, tid, wave, row, hi, lane, i, j, cur_i, acc);
    // odd iteration: compute buf1, prefetch into buf0
    if (p + 1 < PAIRS_PER_CHUNK)
      pair_iter(p + 2 < PAIRS_PER_CHUNK,
                Xi, Xj1, Vji1, Vij1, xiOff, xjOff0, vjiOff0, vijOff0,
                xb, vb, rowbase, tid, wave, row, hi, lane, i, j, cur_i, acc);
  }

  // reduce over N (16 lanes of each half hold k=0..15 of one batch row)
  #pragma unroll
  for (int q = 0; q < 8; ++q) {
    float s = acc[q];
    #pragma unroll
    for (int off = 8; off >= 1; off >>= 1) s += __shfl_xor(s, off, 16);
    if (row == 0)
      atomicAdd(out + rowbase + wave * 16 + q + 8 * hi, s);
  }
}

// ---------------- launch ----------------
extern "C" void kernel_launch(void* const* d_in, const int* in_sizes, int n_in,
                              void* d_out, int out_size, void* d_ws, size_t ws_size,
                              hipStream_t stream) {
  (void)in_sizes; (void)n_in; (void)out_size; (void)ws_size;
  const float* x = (const float*)d_in[0];   // [4096, 10000]
  const float* w = (const float*)d_in[1];   // [10000, 1]
  const float* v = (const float*)d_in[2];   // [20, 10000, 16]
  float* out = (float*)d_out;               // [4096, 1]

  // workspace: pre-converted, pre-padded bf16 copies (~90.4 MB required)
  __bf16* xb = (__bf16*)d_ws;
  __bf16* vb = xb + XB_ELEMS;

  constexpr int LDS_BYTES = (3 * WG_ROWS * SP + 4 * VB_SLAB) * 2;  // 256 KB
  hipFuncSetAttribute((const void*)ffm_pair_kernel,
                      hipFuncAttributeMaxDynamicSharedMemorySize, LDS_BYTES);

  convert_x_kernel<<<BATCH, 256, 0, stream>>>(x, xb);
  convert_v_kernel<<<NUM_FIELDS * NUM_FIELDS, 256, 0, stream>>>(v, vb);
  ffm_linear_kernel<<<BATCH, 256, 0, stream>>>(x, w, out);
  dim3 grid(ROW_TILES, PAIR_CHUNKS);
  ffm_pair_kernel<<<grid, WG_THREADS, LDS_BYTES, stream>>>(xb, vb, out);
}
